// Decoder_24215025615313
// MI455X (gfx1250) — matrix operand order repaired and checked by reading, not compiled, not run on hardware
//
#include <hip/hip_runtime.h>
#include <hip/hip_bf16.h>
#include <math.h>

// Problem constants
#define BB 64
#define SS 2048
#define UU 1024
#define VV 50257
#define LL 8

typedef __attribute__((ext_vector_type(16))) __bf16 v16bf;
typedef __attribute__((ext_vector_type(8)))  __bf16 bf16x8;
typedef __attribute__((ext_vector_type(4)))  __bf16 bf16x4;
typedef __attribute__((ext_vector_type(8)))  float  v8f;

static __device__ inline v8f wmma_bf16(v16bf a, v16bf b, v8f c) {
  return __builtin_amdgcn_wmma_f32_16x16x32_bf16(
      /*neg_a=*/false, a, /*neg_b=*/false, b,
      /*c_mod=*/(short)0, c, /*reuse_a=*/false, /*reuse_b=*/false);
}

// Branch-free fast activations: v_exp_f32 + v_rcp_f32, no EXEC divergence.
static __device__ inline float fast_rcp(float x) {
  return __builtin_amdgcn_rcpf(x);
}
static __device__ inline float fast_tanh(float x) {
  return 1.f - 2.f * fast_rcp(__expf(2.f * x) + 1.f);
}
static __device__ inline float fast_sigmoid(float x) {
  return fast_rcp(1.f + __expf(-x));
}

// A-fragment (16x32 bf16, M x K). Lane l: m = l&15, half kh = l>>4.
// Element e -> K = kh*16+e. src = row K-tile base.
static __device__ inline v16bf load_a_frag(const float* src, int kh) {
  v16bf a;
#pragma unroll
  for (int e = 0; e < 8; ++e) a[e] = (__bf16)src[kh * 16 + e];
#pragma unroll
  for (int e = 0; e < 8; ++e) a[8 + e] = (__bf16)src[kh * 16 + 8 + e];
  return a;
}

// B-fragment from a column-major (transposed) bf16 LDS panel: the lane's 16
// K-values are contiguous -> exactly 2x ds_load_b128.
// base = &BsT[col * LDB + k0 + kh*16]
static __device__ inline v16bf load_b_frag_T(const __bf16* base) {
  bf16x8 lo = *(const bf16x8*)base;
  bf16x8 hi = *(const bf16x8*)(base + 8);
  v16bf b;
#pragma unroll
  for (int e = 0; e < 8; ++e) { b[e] = lo[e]; b[8 + e] = hi[e]; }
  return b;
}

// B-fragment (32x16, K x N) from fp32 k-major LDS tile [k][ldn] (TDM panels).
static __device__ inline v16bf load_b_frag_lds_f32(const float* bs, int ldn,
                                                   int kh, int col) {
  v16bf b;
#pragma unroll
  for (int e = 0; e < 16; ++e) b[e] = (__bf16)bs[(kh * 16 + e) * ldn + col];
  return b;
}

// ---------------------------------------------------------------------------
// Tensor Data Mover: 2D fp32 tile -> LDS (D# per CDNA5 ISA
// §8). One wave issues, waits TENSORcnt, then the WG barrier publishes LDS.
// ---------------------------------------------------------------------------
#if __has_builtin(__builtin_amdgcn_tensor_load_to_lds)
#define HAVE_TDM 1
typedef __attribute__((ext_vector_type(4))) unsigned int u32x4;
typedef __attribute__((ext_vector_type(8))) int i32x8;
typedef __attribute__((ext_vector_type(4))) int i32x4;

static __device__ inline void tdm_load_2d_f32(
    unsigned lds_byte_addr, const float* gaddr,
    unsigned tensor_w,        // tensor_dim0 (elems; OOB beyond -> zero-fill)
    unsigned tile_w,          // tile_dim0 (elems)
    unsigned tile_h,          // tile_dim1 (rows) == tensor_dim1
    unsigned long long row_stride,  // tensor_dim0_stride (elems)
    unsigned pad_interval_code, unsigned pad_amount_code, unsigned pad_en) {
  unsigned long long ga = (unsigned long long)(size_t)gaddr;
  u32x4 g0;
  g0[0] = 1u;                                          // count=1, user mode
  g0[1] = lds_byte_addr;                               // lds_addr
  g0[2] = (unsigned)(ga & 0xffffffffu);                // global_addr[31:0]
  g0[3] = (unsigned)((ga >> 32) & 0x01ffffffu) | (2u << 30);   // [56:32]|type=2
  i32x8 g1;
  g1[0] = (int)((2u << 16)                             // data_size = 4B
                | ((pad_en & 1u) << 20)
                | ((pad_interval_code & 7u) << 22)
                | ((pad_amount_code & 127u) << 25));
  g1[1] = (int)((tensor_w & 0xffffu) << 16);           // tensor_dim0[15:0]
  g1[2] = (int)(((tensor_w >> 16) & 0xffffu)           // tensor_dim0[31:16]
                | ((tile_h & 0xffffu) << 16));         // tensor_dim1[15:0]
  g1[3] = (int)(((tile_h >> 16) & 0xffffu)             // tensor_dim1[31:16]
                | ((tile_w & 0xffffu) << 16));         // tile_dim0
  g1[4] = (int)(tile_h & 0xffffu);                     // tile_dim1; tile_dim2=0
  g1[5] = (int)(unsigned)(row_stride & 0xffffffffu);   // dim0_stride[31:0]
  g1[6] = (int)(unsigned)((row_stride >> 32) & 0xffffu); // dim0_stride[47:32]
  g1[7] = 0;
  i32x4 gz = {0, 0, 0, 0};
#if defined(__clang_major__) && __clang_major__ >= 23
  i32x8 gz8 = {0, 0, 0, 0, 0, 0, 0, 0};
  __builtin_amdgcn_tensor_load_to_lds(g0, g1, gz, gz, gz8, 0);
#else
  __builtin_amdgcn_tensor_load_to_lds(g0, g1, gz, gz, 0);
#endif
}
#else
#define HAVE_TDM 0
#endif

// ---------------------------------------------------------------------------
// 1) qproj[b,n] = query_h @ attn_W2 + attn_b2           (64 x 1024, trivial)
// ---------------------------------------------------------------------------
__global__ void qproj_kernel(const float* __restrict__ qh,
                             const float* __restrict__ W2,
                             const float* __restrict__ b2,
                             float* __restrict__ qproj) {
  int idx = blockIdx.x * 256 + threadIdx.x;     // 65536 total
  int b = idx >> 10, n = idx & 1023;
  float acc = b2[n];
  const float* q = qh + b * UU;
  for (int k = 0; k < UU; ++k) acc = fmaf(q[k], W2[(size_t)k * UU + n], acc);
  qproj[idx] = acc;
}

// ---------------------------------------------------------------------------
// 2) score[b,s] = tanh(enc@W1 + b1 + qproj[b]) . attn_V + bV
//    Fused: the 512MB 'feat' tensor is never written. 128 rows / WG (8 waves
//    x 16 rows). W1 column-panel is staged TRANSPOSED (column-major bf16,
//    row stride 1032) so a B-fragment is 2x ds_load_b128, bank-conflict-free.
// ---------------------------------------------------------------------------
#define LDB (UU + 8)                             // padded col stride (bf16)
__global__ void __launch_bounds__(256) attn_score_kernel(
    const float* __restrict__ enc, const float* __restrict__ W1,
    const float* __restrict__ b1, const float* __restrict__ qproj,
    const float* __restrict__ aV, const float* __restrict__ abV,
    float* __restrict__ score) {
  __shared__ __align__(16) __bf16 BsT[16 * LDB]; // 33KB: W1[:, n0..n0+15]^T
  const int tid = threadIdx.x;
  const int lane = tid & 31, wave = tid >> 5;
  const int kh = lane >> 4, ln = lane & 15;
  const int row0 = blockIdx.x * 128;             // 1024 blocks, S%128==0
  const int b = row0 >> 11;                      // row0 / S
  const int row = row0 + wave * 16 + ln;         // this lane's A row
  const float* arow = enc + (size_t)row * UU;

  float sc[8];
#pragma unroll
  for (int r = 0; r < 8; ++r) sc[r] = 0.f;

  for (int nt = 0; nt < UU / 16; ++nt) {
    const int n0 = nt * 16;
    __syncthreads();
    // Transposing stage: lanes 0..15 read 16 consecutive columns (64B
    // coalesced), 4 consecutive k each; pack and ds_store_b64 contiguous.
    for (int i = tid; i < 16 * (UU / 4); i += 256) {
      const int cc = i & 15;                     // column within panel
      const int k = (i >> 4) * 4;                // k quad
      const float* wp = W1 + (size_t)k * UU + n0 + cc;
      bf16x4 p;
      p[0] = (__bf16)wp[0];
      p[1] = (__bf16)wp[UU];
      p[2] = (__bf16)wp[2 * UU];
      p[3] = (__bf16)wp[3 * UU];
      *(bf16x4*)(BsT + cc * LDB + k) = p;
    }
    __syncthreads();

    v8f c = {};
#pragma unroll 4
    for (int kt = 0; kt < UU / 32; ++kt) {
      const int k0 = kt * 32;
      v16bf a  = load_a_frag(arow + k0, kh);
      v16bf bb = load_b_frag_T(BsT + ln * LDB + k0 + kh * 16);
      c = wmma_bf16(a, bb, c);
    }
    // Epilogue: lane holds n = n0+ln, rows m = r + 8*kh (within wave tile)
    const int n = n0 + ln;
    const float add = b1[n] + qproj[b * UU + n];
    const float vn = aV[n];
#pragma unroll
    for (int r = 0; r < 8; ++r) sc[r] += fast_tanh(c[r] + add) * vn;
  }
  // Reduce across the 16 n-lanes within each half-wave (xor masks 1..8 stay
  // inside a 16-lane group on wave32).
#pragma unroll
  for (int r = 0; r < 8; ++r) {
    sc[r] += __shfl_xor(sc[r], 8, 32);
    sc[r] += __shfl_xor(sc[r], 4, 32);
    sc[r] += __shfl_xor(sc[r], 2, 32);
    sc[r] += __shfl_xor(sc[r], 1, 32);
  }
  if (ln == 0) {
#pragma unroll
    for (int r = 0; r < 8; ++r)
      score[row0 + wave * 16 + r + 8 * kh] = sc[r] + abV[0];
  }
}

// ---------------------------------------------------------------------------
// 3) per-batch softmax stats over S
// ---------------------------------------------------------------------------
__global__ void softmax_stats_kernel(const float* __restrict__ score,
                                     float* __restrict__ stats) {
  __shared__ float sh[256];
  const int b = blockIdx.x, tid = threadIdx.x;
  const float* s = score + b * SS;
  float m = -3.4e38f;
  for (int i = tid; i < SS; i += 256) m = fmaxf(m, s[i]);
  sh[tid] = m; __syncthreads();
  for (int k = 128; k > 0; k >>= 1) {
    if (tid < k) sh[tid] = fmaxf(sh[tid], sh[tid + k]);
    __syncthreads();
  }
  m = sh[0]; __syncthreads();
  float e = 0.f;
  for (int i = tid; i < SS; i += 256) e += __expf(s[i] - m);
  sh[tid] = e; __syncthreads();
  for (int k = 128; k > 0; k >>= 1) {
    if (tid < k) sh[tid] += sh[tid + k];
    __syncthreads();
  }
  if (tid == 0) { stats[b * 2] = m; stats[b * 2 + 1] = sh[0]; }
}

__global__ void attn_weights_kernel(const float* __restrict__ score,
                                    const float* __restrict__ stats,
                                    float* __restrict__ attn_out) {
  int idx = blockIdx.x * 256 + threadIdx.x;      // B*S = 131072
  int b = idx >> 11;
  float m = stats[b * 2], inv = fast_rcp(stats[b * 2 + 1]);
  attn_out[idx] = __expf(score[idx] - m) * inv;
}

// ---------------------------------------------------------------------------
// 4) ctx[b,u] = sum_s attn_w[b,s] * enc[b,s,u]   (2nd, final pass over enc)
// ---------------------------------------------------------------------------
__global__ void ctx_kernel(const float* __restrict__ enc,
                           const float* __restrict__ attn,
                           float* __restrict__ ctx) {
  const int b = blockIdx.x >> 2;
  const int col = (blockIdx.x & 3) * 256 + threadIdx.x;
  const float* e = enc + (size_t)b * SS * UU + col;
  const float* w = attn + b * SS;
  float acc = 0.f;
  for (int s = 0; s < SS; ++s) acc = fmaf(w[s], e[(size_t)s * UU], acc);
  ctx[b * UU + col] = acc;
}

// ---------------------------------------------------------------------------
// 5) one LSTM layer: z = [x,ctx] @ W[i] + b[i]; c=sig(i)*tanh(g);
//    h=sig(o)*tanh(c); x_next = h (+x if residual). One WG per 16-wide u-tile
//    computes all four gate columns so the gate math fuses in-kernel.
//    8 waves = 4 m-tiles x 2 gate-pairs. Weight panels (16x32 fp32, row
//    stride 4096) are DMA'd by the TDM: pad_interval=16 DWORDs, pad_amount=48
//    DWORDs lands rows on the 64-float LDS stride.
// ---------------------------------------------------------------------------
__global__ void __launch_bounds__(256) lstm_layer_kernel(
    const float* __restrict__ xin, const float* __restrict__ ctx,
    const float* __restrict__ W, const float* __restrict__ bias,
    float* __restrict__ hs, float* __restrict__ cs, float* __restrict__ xout,
    int residual) {
  __shared__ __align__(16) float Wsf[32 * 64];   // K-chunk 32 x 64 cols, 8KB
  __shared__ float G[4][64][16];                 // gates for all 64 rows, 16KB
  const int tid = threadIdx.x;
  const int lane = tid & 31, wave = tid >> 5;
  const int kh = lane >> 4, ln = lane & 15;
  const int u0 = blockIdx.x * 16;                // 64 blocks
  const int mtile = wave >> 1;
  const int g0 = (wave & 1) * 2;                 // this wave: gates g0, g0+1
  const int row = mtile * 16 + ln;
#if HAVE_TDM
  const unsigned ws_lds = (unsigned)(size_t)(void*)Wsf;  // LDS byte offset
#endif

  v8f c0 = {}, c1 = {};
  for (int kt = 0; kt < (2 * UU) / 32; ++kt) {
    const int k0 = kt * 32;
    __syncthreads();
#if HAVE_TDM
    if (wave == 0) {
#pragma unroll
      for (int g = 0; g < 4; ++g)
        tdm_load_2d_f32(ws_lds + (unsigned)(g * 16 * 4),
                        W + (size_t)k0 * (4 * UU) + (size_t)g * UU + u0,
                        /*tensor_w=*/16, /*tile_w=*/16, /*tile_h=*/32,
                        /*row_stride=*/4 * UU,
                        /*pad_interval(16dw)=*/3, /*pad_amount(48dw)=*/47, 1);
      __builtin_amdgcn_s_wait_tensorcnt(0);
    }
#else
    for (int i = tid; i < 2048; i += 256) {      // fallback: coop stage
      int k = i >> 6, cg = i & 63, g = cg >> 4, n = cg & 15;
      Wsf[i] = W[(size_t)(k0 + k) * (4 * UU) + g * UU + u0 + n];
    }
#endif
    __syncthreads();
    // inp = concat(x, ctx): K-tile lies entirely in one half (32 | 1024)
    const float* src = (k0 < UU) ? (xin + (size_t)row * UU + k0)
                                 : (ctx + (size_t)row * UU + (k0 - UU));
    v16bf a   = load_a_frag(src, kh);
    v16bf bb0 = load_b_frag_lds_f32(Wsf, 64, kh, g0 * 16 + ln);
    v16bf bb1 = load_b_frag_lds_f32(Wsf, 64, kh, (g0 + 1) * 16 + ln);
    c0 = wmma_bf16(a, bb0, c0);
    c1 = wmma_bf16(a, bb1, c1);
  }
  __syncthreads();
  {
    const float bv0 = bias[g0 * UU + u0 + ln];
    const float bv1 = bias[(g0 + 1) * UU + u0 + ln];
#pragma unroll
    for (int r = 0; r < 8; ++r) {
      int m = mtile * 16 + r + 8 * kh;
      G[g0][m][ln]     = c0[r] + bv0;
      G[g0 + 1][m][ln] = c1[r] + bv1;
    }
  }
  __syncthreads();
  for (int i = tid; i < 64 * 16; i += 256) {
    int r = i >> 4, n = i & 15;
    float ig = G[0][r][n], gg = G[2][r][n], og = G[3][r][n];
    float cv = fast_sigmoid(ig) * fast_tanh(gg);
    float hv = fast_sigmoid(og) * fast_tanh(cv);
    size_t o = (size_t)r * UU + u0 + n;
    hs[o] = hv;
    cs[o] = cv;
    xout[o] = hv + (residual ? xin[o] : 0.f);
  }
}

// ---------------------------------------------------------------------------
// 6) logits = x @ dense_W + b   (64 x 50257, K=1024). 128-col panel per WG;
//    the 206MB dense_W streams through the TDM (tensor_dim0 = V-n0 gives
//    zero-fill on the ragged last panel), WMMA bf16 main loop.
// ---------------------------------------------------------------------------
__global__ void __launch_bounds__(256) dense_kernel(
    const float* __restrict__ x, const float* __restrict__ W,
    const float* __restrict__ bias, float* __restrict__ logits) {
  __shared__ __align__(16) float Ds[32 * 128];   // 16KB fp32 panel
  const int tid = threadIdx.x;
  const int lane = tid & 31, wave = tid >> 5;
  const int kh = lane >> 4, ln = lane & 15;
  const int n0 = blockIdx.x * 128;               // 393 blocks
#if HAVE_TDM
  const unsigned ds_lds = (unsigned)(size_t)(void*)Ds;
#endif

  v8f c[4];
#pragma unroll
  for (int mt = 0; mt < 4; ++mt) c[mt] = (v8f){};

  for (int kt = 0; kt < UU / 32; ++kt) {
    const int k0 = kt * 32;
    __syncthreads();
#if HAVE_TDM
    if (wave == 0) {
      tdm_load_2d_f32(ds_lds, W + (size_t)k0 * VV + n0,
                      /*tensor_w=*/(unsigned)(VV - n0), /*tile_w=*/128,
                      /*tile_h=*/32, /*row_stride=*/VV, 0, 0, 0);
      __builtin_amdgcn_s_wait_tensorcnt(0);
    }
#else
    for (int i = tid; i < 4096; i += 256) {
      int k = i >> 7, cc = i & 127, n = n0 + cc;
      Ds[i] = (n < VV) ? W[(size_t)(k0 + k) * VV + n] : 0.f;
    }
#endif
    __syncthreads();
    v16bf bb = load_b_frag_lds_f32(Ds, 128, kh, wave * 16 + ln);
#pragma unroll
    for (int mt = 0; mt < 4; ++mt) {
      v16bf a = load_a_frag(x + (size_t)(mt * 16 + ln) * UU + k0, kh);
      c[mt] = wmma_bf16(a, bb, c[mt]);
    }
  }
  const int n = n0 + wave * 16 + ln;
  if (n < VV) {
    const float bv = bias[n];
#pragma unroll
    for (int mt = 0; mt < 4; ++mt)
#pragma unroll
      for (int r = 0; r < 8; ++r)
        logits[(size_t)(mt * 16 + r + 8 * kh) * VV + n] = c[mt][r] + bv;
  }
}

// ---------------------------------------------------------------------------
// 7) probs = softmax(logits) over V
// ---------------------------------------------------------------------------
__global__ void probs_kernel(const float* __restrict__ logits,
                             float* __restrict__ probs) {
  __shared__ float sh[256];
  const int b = blockIdx.x, tid = threadIdx.x;
  const float* l = logits + (size_t)b * VV;
  float m = -3.4e38f;
  for (int i = tid; i < VV; i += 256) m = fmaxf(m, l[i]);
  sh[tid] = m; __syncthreads();
  for (int k = 128; k > 0; k >>= 1) {
    if (tid < k) sh[tid] = fmaxf(sh[tid], sh[tid + k]);
    __syncthreads();
  }
  m = sh[0]; __syncthreads();
  float e = 0.f;
  for (int i = tid; i < VV; i += 256) e += __expf(l[i] - m);
  sh[tid] = e; __syncthreads();
  for (int k = 128; k > 0; k >>= 1) {
    if (tid < k) sh[tid] += sh[tid + k];
    __syncthreads();
  }
  const float inv = fast_rcp(sh[0]);
  for (int i = tid; i < VV; i += 256)
    probs[(size_t)b * VV + i] = __expf(l[i] - m) * inv;
}

// ---------------------------------------------------------------------------
extern "C" void kernel_launch(void* const* d_in, const int* in_sizes, int n_in,
                              void* d_out, int out_size, void* d_ws,
                              size_t ws_size, hipStream_t stream) {
  const float* enc = (const float*)d_in[0];   // B,S,U
  const float* dec = (const float*)d_in[1];   // B,1,U
  const float* qh  = (const float*)d_in[2];   // B,U
  const float* W1  = (const float*)d_in[3];   // U,U
  const float* b1  = (const float*)d_in[4];   // U
  const float* W2  = (const float*)d_in[5];   // U,U
  const float* b2  = (const float*)d_in[6];   // U
  const float* aV  = (const float*)d_in[7];   // U,1
  const float* abV = (const float*)d_in[8];   // 1
  const float* lW  = (const float*)d_in[9];   // L,2U,4U
  const float* lb  = (const float*)d_in[10];  // L,4U
  const float* dW  = (const float*)d_in[11];  // U,V
  const float* db  = (const float*)d_in[12];  // V

  float* out = (float*)d_out;
  float* probs    = out;                              // B*V
  float* hs_out   = out + (size_t)BB * VV;            // L*B*U
  float* cs_out   = hs_out + (size_t)LL * BB * UU;    // L*B*U
  float* attn_out = cs_out + (size_t)LL * BB * UU;    // B*S

  float* ws = (float*)d_ws;
  float* qproj  = ws;                      // 65536
  float* score  = qproj + BB * UU;         // 131072
  float* stats  = score + BB * SS;         // 128
  float* ctx    = stats + 2 * BB;          // 65536
  float* xbuf   = ctx + BB * UU;           // 2 * 65536 (ping/pong)
  float* logits = xbuf + 2 * BB * UU;      // B*V (~12.9MB)

  qproj_kernel<<<(BB * UU) / 256, 256, 0, stream>>>(qh, W2, b2, qproj);
  attn_score_kernel<<<(BB * SS) / 128, 256, 0, stream>>>(enc, W1, b1, qproj,
                                                         aV, abV, score);
  softmax_stats_kernel<<<BB, 256, 0, stream>>>(score, stats);
  attn_weights_kernel<<<(BB * SS) / 256, 256, 0, stream>>>(score, stats,
                                                           attn_out);
  ctx_kernel<<<BB * 4, 256, 0, stream>>>(enc, attn_out, ctx);

  static const int kResidual[LL] = {0, 0, 1, 1, 1, 1, 1, 0};
  const float* xin = dec;
  for (int i = 0; i < LL; ++i) {
    float* xout = xbuf + (i & 1) * BB * UU;
    lstm_layer_kernel<<<UU / 16, 256, 0, stream>>>(
        xin, ctx, lW + (size_t)i * (2 * UU) * (4 * UU), lb + i * (4 * UU),
        hs_out + (size_t)i * BB * UU, cs_out + (size_t)i * BB * UU, xout,
        kResidual[i]);
    xin = xout;
  }
  dense_kernel<<<(VV + 127) / 128, 256, 0, stream>>>(xin, dW, db, logits);
  probs_kernel<<<BB, 256, 0, stream>>>(logits, probs);
}
